// Refine_81956565942273
// MI455X (gfx1250) — matrix operand order repaired and checked by reading, not compiled, not run on hardware
//
#include <hip/hip_runtime.h>
#include <hip/hip_bf16.h>
#include <cstdint>

// ---------------------------------------------------------------------------
// Problem constants (reference: B=8, C=1024, H=W=64, K=64)
// ---------------------------------------------------------------------------
#define BATCH 8
#define CH    1024
#define HW    4096          // 64*64 pixels per batch image
#define NK    64            // number of centroids

typedef __attribute__((ext_vector_type(16))) _Float16 v16h;
typedef __attribute__((ext_vector_type(8)))  _Float16 v8h;
typedef __attribute__((ext_vector_type(8)))  float    v8f;

#define WMMA_F16(A, B, Cacc) \
  __builtin_amdgcn_wmma_f32_16x16x32_f16(false, (A), false, (B), (short)0, (Cacc), false, false)

#define BPAD 40   // padded LDS row stride in halves (80B -> 16 distinct banks)

// ---------------------------------------------------------------------------
// Async global->LDS copy (CDNA5 GLOBAL_LOAD_ASYNC_TO_LDS_B128, ASYNCcnt).
// LDS address operand = low 32 bits of the generic shared-mem pointer
// ---------------------------------------------------------------------------
__device__ __forceinline__ void async_copy_b128(void* lds_dst, const void* gsrc) {
  uint32_t lo = (uint32_t)(uintptr_t)lds_dst;
  asm volatile("global_load_async_to_lds_b128 %0, %1, off"
               :: "v"(lo), "v"(gsrc) : "memory");
}
__device__ __forceinline__ void wait_async0() {
  asm volatile("s_wait_asynccnt 0" ::: "memory");
}

// stage one 64-row x 32-half tile: each thread copies 32B (two b128 chunks)
__device__ __forceinline__ void stage_tile(_Float16 (*buf)[BPAD], const _Float16* src,
                                           int trow, int thalf) {
  async_copy_b128(&buf[trow][thalf],     src);
  async_copy_b128(&buf[trow][thalf + 8], src + 8);
}

// ---------------------------------------------------------------------------
// WMMA fragment loaders (contiguous k order, same for A and B, wave32)
// A: 16x32 f16, row m = lane&15; halves 0..15 -> K = kb..kb+15 (kb = 16*(lane>>4))
// B: 32x16 f16, col n = lane&15; halves 0..15 -> K = koff..koff+15 (koff = 16*(lane>>4))
// ---------------------------------------------------------------------------
__device__ __forceinline__ v16h load_frag_a(const _Float16* base, int ld, int row0, int c0) {
  const int lane = threadIdx.x & 31;
  const int r    = row0 + (lane & 15);
  const int kb   = (lane >> 4) * 16;
  const _Float16* p = base + r * ld + c0;
  return *(const v16h*)(p + kb);
}

// 4-tile WMMA group: preload ALL fragments (8 ds_load_b128 clause, one dscnt
// wait), then issue the 4 WMMAs back-to-back.
__device__ __forceinline__ void wmma4(const _Float16 (*buf)[BPAD], const v16h& a,
                                      v8f* acc, int ln, int koff) {
  v16h bf[4];
#pragma unroll
  for (int j = 0; j < 4; ++j) bf[j] = *(const v16h*)&buf[j * 16 + ln][koff];
#pragma unroll
  for (int j = 0; j < 4; ++j) acc[j] = WMMA_F16(a, bf[j], acc[j]);
}

// 8-tile WMMA group (two A fragments sharing the 4 preloaded B fragments)
__device__ __forceinline__ void wmma8(const _Float16 (*buf)[BPAD], const v16h& a0, const v16h& a1,
                                      v8f (*acc)[4], int ln, int koff) {
  v16h bf[4];
#pragma unroll
  for (int j = 0; j < 4; ++j) bf[j] = *(const v16h*)&buf[j * 16 + ln][koff];
#pragma unroll
  for (int j = 0; j < 4; ++j) {
    acc[0][j] = WMMA_F16(a0, bf[j], acc[0][j]);
    acc[1][j] = WMMA_F16(a1, bf[j], acc[1][j]);
  }
}

// ---------------------------------------------------------------------------
// K0a: normalize centroids along C, convert to f16   gh[k][c] (row-major)
// ---------------------------------------------------------------------------
__global__ void k_norm_centroids(const float* __restrict__ cent, _Float16* __restrict__ gh) {
  __shared__ float red[256];
  const int k = blockIdx.x;
  const int t = threadIdx.x;
  float ss = 0.f;
  for (int c = t; c < CH; c += 256) { float v = cent[k * CH + c]; ss += v * v; }
  red[t] = ss; __syncthreads();
  for (int s = 128; s > 0; s >>= 1) { if (t < s) red[t] += red[t + s]; __syncthreads(); }
  const float inv = 1.f / fmaxf(sqrtf(red[0]), 1e-12f);
  for (int c = t; c < CH; c += 256)
    gh[k * CH + c] = (_Float16)(cent[k * CH + c] * inv);
}

// K0c: fc_weight f32 -> f16
__global__ void k_cvt_weight(const float* __restrict__ wsrc, _Float16* __restrict__ wh) {
  const int i = blockIdx.x * 256 + threadIdx.x;
  wh[i] = (_Float16)wsrc[i];
}

// ---------------------------------------------------------------------------
// K0b: transpose-convert x [b][c][p] f32 -> xhT [b][p][c] f16  (LDS-tiled)
// ---------------------------------------------------------------------------
__global__ void k_transpose_x(const float* __restrict__ x, _Float16* __restrict__ xhT) {
  __shared__ _Float16 st[64][65];
  const int b  = blockIdx.x >> 6;
  const int p0 = (blockIdx.x & 63) * 64;
  const int t  = threadIdx.x;
  const int q  = t >> 6;          // 0..3
  const int pp = t & 63;          // 0..63
  for (int ct = 0; ct < 16; ++ct) {
#pragma unroll
    for (int i = 0; i < 16; ++i) {
      const int ci = i * 4 + q;
      st[ci][pp] = (_Float16)x[((size_t)(b * CH + ct * 64 + ci)) * HW + p0 + pp];
    }
    __syncthreads();
#pragma unroll
    for (int i = 0; i < 16; ++i) {
      const int pi = i * 4 + q;
      xhT[((size_t)(b * HW + p0 + pi)) * CH + ct * 64 + pp] = st[pp][pi];
    }
    __syncthreads();
  }
}

// ---------------------------------------------------------------------------
// K1: sim GEMM + argmax. Block = 4 waves (each 16 pixels x all 64 centroids).
// Centroid tiles async-staged into LDS ping/pong buffers; explicit 2-phase
// loop over 64-channel super-steps (even, symmetric bodies -> in-place acc).
// ---------------------------------------------------------------------------
__device__ __forceinline__ void amax2(float& v, int& n, float v2, int n2) {
  if (v2 > v || (v2 == v && n2 < n)) { v = v2; n = n2; }
}

__global__ void k_assign(const _Float16* __restrict__ xhT, const _Float16* __restrict__ gh,
                         int* __restrict__ idxb) {
  __shared__ __align__(16) _Float16 gs[2][64][BPAD];
  const int gw = blockIdx.x * 4 + (threadIdx.x >> 5);
  const int b  = gw >> 8;                 // 256 pixel-tiles per batch
  const int p0 = (gw & 255) * 16;
  const _Float16* xb = xhT + (size_t)b * HW * CH;

  const int trow  = threadIdx.x >> 1;
  const int thalf = (threadIdx.x & 1) * 16;
  const _Float16* gsrc = gh + (size_t)trow * CH + thalf;

  const int lane = threadIdx.x & 31;
  const int ln = lane & 15, hi = lane >> 4;
  const int koff = hi * 16;

  // prologue: buffer 0 <- K-step 0
  stage_tile(gs[0], gsrc, trow, thalf);
  v16h a = load_frag_a(xb, CH, p0, 0);
  wait_async0();
  __syncthreads();

  v8f acc[4] = {};
  for (int c0 = 0; c0 < CH - 64; c0 += 64) {
    // phase A: compute on gs[0] (K=c0), stage gs[1] <- c0+32
    stage_tile(gs[1], gsrc + c0 + 32, trow, thalf);
    v16h a1 = load_frag_a(xb, CH, p0, c0 + 32);
    wmma4(gs[0], a, acc, ln, koff);
    wait_async0();
    __syncthreads();
    // phase B: compute on gs[1] (K=c0+32), stage gs[0] <- c0+64
    stage_tile(gs[0], gsrc + c0 + 64, trow, thalf);
    a = load_frag_a(xb, CH, p0, c0 + 64);
    wmma4(gs[1], a1, acc, ln, koff);
    wait_async0();
    __syncthreads();
  }
  // epilogue: gs[0] holds CH-64; stage gs[1] <- CH-32
  stage_tile(gs[1], gsrc + CH - 32, trow, thalf);
  v16h alast = load_frag_a(xb, CH, p0, CH - 32);
  wmma4(gs[0], a, acc, ln, koff);
  wait_async0();
  __syncthreads();
  wmma4(gs[1], alast, acc, ln, koff);

  // argmax over 64 centroids (4 tiles x 16 lanes), first-max tie-break
#pragma unroll
  for (int r = 0; r < 8; ++r) {
    float bv = acc[0][r]; int bn2 = ln;
    amax2(bv, bn2, acc[1][r], 16 + ln);
    amax2(bv, bn2, acc[2][r], 32 + ln);
    amax2(bv, bn2, acc[3][r], 48 + ln);
#pragma unroll
    for (int m = 8; m >= 1; m >>= 1) {
      float ov = __shfl_xor(bv, m, 16);
      int   on = __shfl_xor(bn2, m, 16);
      amax2(bv, bn2, ov, on);
    }
    if (ln == 0) idxb[b * HW + p0 + r + 8 * hi] = bn2;  // D row m = r + 8*hi
  }
}

// ---------------------------------------------------------------------------
// K2: per-batch segmented sums: sumx[b][k][c], count[b][k] via LDS f32 atomics
// ---------------------------------------------------------------------------
__global__ void k_stats(const float* __restrict__ x, const int* __restrict__ idxb,
                        float* __restrict__ sumx, float* __restrict__ cntb) {
  __shared__ float acc[NK * 129];
  __shared__ float cnt[NK];
  const int b  = blockIdx.x >> 3;
  const int c0 = (blockIdx.x & 7) * 128;
  const int t  = threadIdx.x;
  for (int e = t; e < NK * 129; e += 256) acc[e] = 0.f;
  if (t < NK) cnt[t] = 0.f;
  __syncthreads();

  for (int pb = 0; pb < HW; pb += 256) {
    const int p = pb + t;
    const int k = idxb[b * HW + p] & 63;
    if (c0 == 0) atomicAdd(&cnt[k], 1.f);
    if (pb + 256 < HW) {   // pull next pixel chunk toward the caches early
      __builtin_prefetch(&x[((size_t)(b * CH + c0)) * HW + p + 256], 0, 1);
      __builtin_prefetch(&x[((size_t)(b * CH + c0 + 64)) * HW + p + 256], 0, 1);
    }
#pragma unroll 4
    for (int ci = 0; ci < 128; ++ci)
      atomicAdd(&acc[k * 129 + ci], x[((size_t)(b * CH + c0 + ci)) * HW + p]);
  }
  __syncthreads();
  for (int e = t; e < NK * 128; e += 256) {
    const int k = e >> 7, ci = e & 127;
    sumx[((size_t)(b * NK + k)) * CH + c0 + ci] = acc[k * 129 + ci];
  }
  if (c0 == 0 && t < NK) cntb[b * NK + t] = cnt[t];
}

// K2b: centroids_local = sumx / max(count,1)   (in place)
__global__ void k_finalize(float* __restrict__ sumx, const float* __restrict__ cntb) {
  const int i  = blockIdx.x * 256 + threadIdx.x;   // over 8*64*1024
  const int bk = i >> 10;
  sumx[i] /= fmaxf(cntb[bk], 1.f);
}

// ---------------------------------------------------------------------------
// K3: fused calibration: w = exp(-mean_c(delta^2)); x_cal = x + w*delta,
// written transposed as f16 into xhT (reuse) -> conv B operand.
// ---------------------------------------------------------------------------
__global__ void k_calibrate(const float* __restrict__ x, const int* __restrict__ idxb,
                            const float* __restrict__ cl, _Float16* __restrict__ xcT) {
  __shared__ float red[256];
  __shared__ float wsh[64];
  __shared__ _Float16 st[64][65];
  const int b  = blockIdx.x >> 6;
  const int p0 = (blockIdx.x & 63) * 64;
  const int t  = threadIdx.x;
  const int q  = t >> 6;
  const int pp = t & 63;
  const int p  = p0 + pp;
  const int k  = idxb[b * HW + p] & 63;
  const float* clrow = cl + ((size_t)(b * NK + k)) * CH;
  const float* xcol  = x + (size_t)b * CH * HW + p;

  float part = 0.f;
  for (int c = q; c < CH; c += 4) {
    const float d = clrow[c] - xcol[(size_t)c * HW];
    part += d * d;
  }
  red[t] = part; __syncthreads();
  if (t < 64) wsh[t] = __expf(-(red[t] + red[64 + t] + red[128 + t] + red[192 + t]) * (1.f / CH));
  __syncthreads();
  const float wv = wsh[pp];

  for (int ct = 0; ct < 16; ++ct) {
#pragma unroll
    for (int i = 0; i < 16; ++i) {
      const int c  = ct * 64 + i * 4 + q;
      const float xa = xcol[(size_t)c * HW];
      const float d  = clrow[c] - xa;
      st[i * 4 + q][pp] = (_Float16)(xa + wv * d);
    }
    __syncthreads();
#pragma unroll
    for (int i = 0; i < 16; ++i) {
      const int pi = i * 4 + q;
      xcT[((size_t)(b * HW + p0 + pi)) * CH + ct * 64 + pp] = st[pp][pi];
    }
    __syncthreads();
  }
}

// ---------------------------------------------------------------------------
// K4: 1x1 conv GEMM: out[b][o][p] = relu(sum_c wh[o][c]*xcT[b][p][c] + bias[o])
// Block = 4 waves sharing one 64-pixel panel, covering 128 output channels.
// B tiles async-staged into LDS ping/pong buffers; A (weights) double-buffered
// in registers; explicit 2-phase loop over 64-channel super-steps.
// ---------------------------------------------------------------------------
__global__ void k_conv(const _Float16* __restrict__ wh, const _Float16* __restrict__ xcT,
                       const float* __restrict__ bias, float* __restrict__ outp) {
  __shared__ __align__(16) _Float16 bs[2][64][BPAD];
  const int b      = blockIdx.x >> 9;          // 512 blocks per batch
  const int rem    = blockIdx.x & 511;
  const int o_base = (rem >> 6) * 128;         // 8 o-blocks of 128 channels
  const int p0     = (rem & 63) * 64;          // 64 pixel panels
  const int wv     = threadIdx.x >> 5;         // wave in block: 0..3
  const int o0     = o_base + wv * 32;
  const _Float16* xb = xcT + (size_t)b * HW * CH;

  const int trow  = threadIdx.x >> 1;
  const int thalf = (threadIdx.x & 1) * 16;
  const _Float16* bsrc = xb + (size_t)(p0 + trow) * CH + thalf;

  const int lane = threadIdx.x & 31;
  const int ln = lane & 15, hi = lane >> 4;
  const int koff = hi * 16;

  // prologue: buffer 0 <- K-step 0, first A fragments
  stage_tile(bs[0], bsrc, trow, thalf);
  v16h a0 = load_frag_a(wh, CH, o0,      0);
  v16h a1 = load_frag_a(wh, CH, o0 + 16, 0);
  wait_async0();
  __syncthreads();

  v8f acc[2][4] = {};
  for (int c0 = 0; c0 < CH - 64; c0 += 64) {
    // phase A: compute on bs[0] (K=c0), stage bs[1] <- c0+32
    stage_tile(bs[1], bsrc + c0 + 32, trow, thalf);
    v16h a0n = load_frag_a(wh, CH, o0,      c0 + 32);
    v16h a1n = load_frag_a(wh, CH, o0 + 16, c0 + 32);
    wmma8(bs[0], a0, a1, acc, ln, koff);
    wait_async0();
    __syncthreads();
    // phase B: compute on bs[1] (K=c0+32), stage bs[0] <- c0+64
    stage_tile(bs[0], bsrc + c0 + 64, trow, thalf);
    a0 = load_frag_a(wh, CH, o0,      c0 + 64);
    a1 = load_frag_a(wh, CH, o0 + 16, c0 + 64);
    wmma8(bs[1], a0n, a1n, acc, ln, koff);
    wait_async0();
    __syncthreads();
  }
  // epilogue: bs[0] holds CH-64; stage bs[1] <- CH-32
  stage_tile(bs[1], bsrc + CH - 32, trow, thalf);
  v16h a0l = load_frag_a(wh, CH, o0,      CH - 32);
  v16h a1l = load_frag_a(wh, CH, o0 + 16, CH - 32);
  wmma8(bs[0], a0, a1, acc, ln, koff);
  wait_async0();
  __syncthreads();
  wmma8(bs[1], a0l, a1l, acc, ln, koff);

  // epilogue: bias + relu, D row m = r + 8*hi, col n = ln
#pragma unroll
  for (int i = 0; i < 2; ++i) {
#pragma unroll
    for (int r = 0; r < 8; ++r) {
      const int o = o0 + i * 16 + r + 8 * hi;
      const float bv = bias[o];
#pragma unroll
      for (int j = 0; j < 4; ++j) {
        float v = acc[i][j][r] + bv;
        outp[((size_t)(b * CH + o)) * HW + p0 + j * 16 + ln] = v > 0.f ? v : 0.f;
      }
    }
  }
}

// ---------------------------------------------------------------------------
// Launch: ws layout (bytes):
//   [0, 64MB)        xhT / xcT  (f16, reused: x transpose -> calibrated x)
//   +128KB           gh   (normalized centroids f16)
//   +2MB             wh   (fc_weight f16)
//   +128KB           idx  (int)
//   +2MB             sumx (f32; becomes centroids_local in place)
//   +2KB             count(f32)
// Total ~71.6 MB
// ---------------------------------------------------------------------------
extern "C" void kernel_launch(void* const* d_in, const int* in_sizes, int n_in,
                              void* d_out, int out_size, void* d_ws, size_t ws_size,
                              hipStream_t stream) {
  const float* x    = (const float*)d_in[0];
  const float* cent = (const float*)d_in[1];
  const float* fw   = (const float*)d_in[2];
  const float* fb   = (const float*)d_in[3];
  float* outp = (float*)d_out;

  char* w = (char*)d_ws;
  _Float16* xhT  = (_Float16*)(w);
  _Float16* gh   = (_Float16*)(w + 67108864);
  _Float16* wh   = (_Float16*)(w + 67108864 + 131072);
  int*      idxb = (int*)     (w + 67108864 + 131072 + 2097152);
  float*    sumx = (float*)   (w + 67108864 + 131072 + 2097152 + 131072);
  float*    cntb = (float*)   (w + 67108864 + 131072 + 2097152 + 131072 + 2097152);

  hipLaunchKernelGGL(k_norm_centroids, dim3(NK),   dim3(256), 0, stream, cent, gh);
  hipLaunchKernelGGL(k_cvt_weight,     dim3(4096), dim3(256), 0, stream, fw, wh);
  hipLaunchKernelGGL(k_transpose_x,    dim3(BATCH * 64), dim3(256), 0, stream, x, xhT);
  hipLaunchKernelGGL(k_assign,         dim3(512),  dim3(128), 0, stream, xhT, gh, idxb);
  hipLaunchKernelGGL(k_stats,          dim3(BATCH * 8), dim3(256), 0, stream, x, idxb, sumx, cntb);
  hipLaunchKernelGGL(k_finalize,       dim3(2048), dim3(256), 0, stream, sumx, cntb);
  hipLaunchKernelGGL(k_calibrate,      dim3(BATCH * 64), dim3(256), 0, stream, x, idxb, sumx, xhT);
  hipLaunchKernelGGL(k_conv,           dim3(4096), dim3(128), 0, stream, wh, xhT, fb, outp);
}
